// HeadExtractor_37409165148775
// MI455X (gfx1250) — matrix operand order repaired and checked by reading, not compiled, not run on hardware
//
#include <hip/hip_runtime.h>
#include <hip/hip_bf16.h>

// ---------------------------------------------------------------------------
// Problem constants (match reference)
// ---------------------------------------------------------------------------
#define BGR   32            // graphs
#define NPG   4000          // nodes per graph
#define SSZ   512           // subset size
#define EPG   8192          // edges per graph
#define DIM   256           // embedding dim
#define NHEAD 4
#define CPH   64            // channels per head
#define HE    256
#define NSEG  (BGR*SSZ)     // 16384
#define E0    (BGR*EPG)     // 262144 graph edges
#define ETOT  (E0+NSEG)     // + self loops = 278528

// ---------------------------------------------------------------------------
// Types for WMMA fragments (trivially-copyable ext vectors -> legal in unions)
// ---------------------------------------------------------------------------
typedef __attribute__((ext_vector_type(16))) __bf16 v16bf;
typedef __attribute__((ext_vector_type(8)))  float  v8f;
typedef __attribute__((ext_vector_type(4)))  unsigned int v4u;
typedef __attribute__((ext_vector_type(4)))  float  v4f;

union Frag16 { v4u q[2]; v16bf v; };

__device__ __forceinline__ unsigned short f2bf(float f) {
    unsigned int u = __float_as_uint(f);
    unsigned int r = u + 0x7FFFu + ((u >> 16) & 1u);   // round-to-nearest-even
    return (unsigned short)(r >> 16);
}
__device__ __forceinline__ unsigned int fkey(float f) {   // order-preserving f32 -> u32
    unsigned int u = __float_as_uint(f);
    return u ^ ((((int)u) >> 31) | 0x80000000u);
}
__device__ __forceinline__ float unfkey(unsigned int k) {
    unsigned int u = (k & 0x80000000u) ? (k ^ 0x80000000u) : ~k;
    return __uint_as_float(u);
}
__device__ __forceinline__ void atomAddF(float* p, float v) {
    __hip_atomic_fetch_add(p, v, __ATOMIC_RELAXED, __HIP_MEMORY_SCOPE_AGENT);
}

// ---------------------------------------------------------------------------
// 0) inv[subset[s]] = s  (inv pre-memset to -1)
// ---------------------------------------------------------------------------
__global__ void k_build_inv(const int* __restrict__ subset, int* __restrict__ inv) {
    int s = threadIdx.x;                 // 512 threads
    inv[subset[s]] = s;
}

// ---------------------------------------------------------------------------
// 1) relabel edges + append self loops
// ---------------------------------------------------------------------------
__global__ void k_relabel(const int* __restrict__ ei, const int* __restrict__ batch,
                          const int* __restrict__ inv,
                          int* __restrict__ srcl, int* __restrict__ dstl) {
    int e = blockIdx.x * 256 + threadIdx.x;
    if (e >= ETOT) return;
    if (e < E0) {
        int s = ei[e], d = ei[E0 + e];
        int gs = batch[s], gd = batch[d];
        srcl[e] = gs * SSZ + inv[s - gs * NPG];
        dstl[e] = gd * SSZ + inv[d - gd * NPG];
    } else {
        int i = e - E0;
        srcl[e] = i; dstl[e] = i;
    }
}

// ---------------------------------------------------------------------------
// 2) gather subset rows, keep f32 master + bf16 copy for WMMA
// ---------------------------------------------------------------------------
__global__ void k_gather(const float* __restrict__ ne, const int* __restrict__ subset,
                         float* __restrict__ x, unsigned short* __restrict__ xb) {
    int idx = blockIdx.x * 256 + threadIdx.x;       // NSEG*DIM total
    int i = idx >> 8, d = idx & 255;
    int g = i >> 9, s = i & 511;
    int gid = g * NPG + subset[s];
    float v = ne[(size_t)gid * DIM + d];
    x[idx] = v;
    xb[idx] = f2bf(v);
}

// ---------------------------------------------------------------------------
// 3) GEMM  out[M,256] = bf16(x)[M,256] @ W[256,256] + bias   (WMMA bf16, f32 acc)
//    block = 256 thr = 8 waves; wave -> 16x64 tile (1 A frag reused by 4 B frags
//    per K-step -> 32 WMMAs/wave); block covers 128 rows x 64-col strip.
//    grid = (M/128, 256/64, 2)   gridDim.z: 0 -> Wl/xl, 1 -> Wr/xr
// ---------------------------------------------------------------------------
__global__ void k_gemm(const unsigned short* __restrict__ xb,
                       const float* __restrict__ Wl, const float* __restrict__ Wr,
                       const float* __restrict__ bl, const float* __restrict__ br,
                       float* __restrict__ xl, float* __restrict__ xr) {
    __shared__ unsigned short Wt[64][264];          // W^T strip [n][k], padded
    const float* W    = blockIdx.z ? Wr : Wl;
    const float* bias = blockIdx.z ? br : bl;
    float*       out  = blockIdx.z ? xr : xl;
    const int n0 = blockIdx.y * 64;

    // stage 64-col weight strip, f32 -> bf16, transposed to [n][k]
    for (int idx = threadIdx.x; idx < 64 * 256; idx += 256) {
        int k = idx >> 6, n = idx & 63;
        Wt[n][k] = f2bf(W[k * 256 + n0 + n]);
    }
    __syncthreads();

    const int wave = threadIdx.x >> 5;
    const int lane = threadIdx.x & 31;
    const int half = lane >> 4;
    const int ln   = lane & 15;
    const int m    = blockIdx.x * 128 + wave * 16 + ln;
    const unsigned short* arow = xb + (size_t)m * 256;

    v8f acc0 = {0.f,0.f,0.f,0.f,0.f,0.f,0.f,0.f};
    v8f acc1 = acc0, acc2 = acc0, acc3 = acc0;

    #pragma unroll
    for (int kk = 0; kk < 8; ++kk) {
        Frag16 a;
        const int ka = kk * 32 + half * 8;          // A: 16-bit layout, K split 8/8
        a.q[0] = *(const v4u*)(arow + ka);
        a.q[1] = *(const v4u*)(arow + ka + 16);
        const int kb = kk * 32 + half * 8;          // B: K split 8/8
        Frag16 b0, b1, b2, b3;
        b0.q[0] = *(const v4u*)(&Wt[ln     ][kb]);
        b0.q[1] = *(const v4u*)(&Wt[ln     ][kb + 16]);
        b1.q[0] = *(const v4u*)(&Wt[ln + 16][kb]);
        b1.q[1] = *(const v4u*)(&Wt[ln + 16][kb + 16]);
        b2.q[0] = *(const v4u*)(&Wt[ln + 32][kb]);
        b2.q[1] = *(const v4u*)(&Wt[ln + 32][kb + 16]);
        b3.q[0] = *(const v4u*)(&Wt[ln + 48][kb]);
        b3.q[1] = *(const v4u*)(&Wt[ln + 48][kb + 16]);
        acc0 = __builtin_amdgcn_wmma_f32_16x16x32_bf16(
                   false, a.v, false, b0.v, (short)0, acc0, false, false);
        acc1 = __builtin_amdgcn_wmma_f32_16x16x32_bf16(
                   false, a.v, false, b1.v, (short)0, acc1, false, false);
        acc2 = __builtin_amdgcn_wmma_f32_16x16x32_bf16(
                   false, a.v, false, b2.v, (short)0, acc2, false, false);
        acc3 = __builtin_amdgcn_wmma_f32_16x16x32_bf16(
                   false, a.v, false, b3.v, (short)0, acc3, false, false);
    }

    const int rbase = blockIdx.x * 128 + wave * 16 + half * 8;
    #pragma unroll
    for (int t = 0; t < 4; ++t) {
        const v8f acc = (t == 0) ? acc0 : (t == 1) ? acc1 : (t == 2) ? acc2 : acc3;
        const int n = n0 + t * 16 + ln;
        const float bv = bias[n];
        #pragma unroll
        for (int r = 0; r < 8; ++r)
            out[(size_t)(rbase + r) * 256 + n] = acc[r] + bv;
    }
}

// ---------------------------------------------------------------------------
// 4a) per-edge logits + segment max (wave per edge; 8 dims/lane; head = lane/8)
// ---------------------------------------------------------------------------
__global__ void k_edge_logit(const float* __restrict__ xl, const float* __restrict__ xr,
                             const float* __restrict__ att,
                             const int* __restrict__ srcl, const int* __restrict__ dstl,
                             float* __restrict__ logit, unsigned int* __restrict__ mxkey) {
    int e = blockIdx.x * 8 + (threadIdx.x >> 5);
    if (e >= ETOT) return;
    int lane = threadIdx.x & 31;
    int s = srcl[e], d = dstl[e];
    int d0 = lane * 8;
    int h  = lane >> 3;
    const v4f* pa = (const v4f*)(xl + (size_t)s * 256 + d0);
    const v4f* pb = (const v4f*)(xr + (size_t)d * 256 + d0);
    const v4f* pt = (const v4f*)(att + h * 64 + ((lane & 7) * 8));
    float part = 0.f;
    #pragma unroll
    for (int q = 0; q < 2; ++q) {
        v4f a = pa[q], b = pb[q], t = pt[q];
        float m0 = a.x + b.x; m0 = m0 > 0.f ? m0 : 0.2f * m0;
        float m1 = a.y + b.y; m1 = m1 > 0.f ? m1 : 0.2f * m1;
        float m2 = a.z + b.z; m2 = m2 > 0.f ? m2 : 0.2f * m2;
        float m3 = a.w + b.w; m3 = m3 > 0.f ? m3 : 0.2f * m3;
        part += m0 * t.x + m1 * t.y + m2 * t.z + m3 * t.w;
    }
    part += __shfl_xor(part, 1);
    part += __shfl_xor(part, 2);
    part += __shfl_xor(part, 4);
    if ((lane & 7) == 0) {
        logit[e * NHEAD + h] = part;
        atomicMax(&mxkey[d * NHEAD + h], fkey(part));
    }
}

// ---------------------------------------------------------------------------
// 4b) exp(logit - max) + segment sum
// ---------------------------------------------------------------------------
__global__ void k_edge_exp(const int* __restrict__ dstl, const unsigned int* __restrict__ mxkey,
                           float* __restrict__ eval, float* __restrict__ den) {
    int idx = blockIdx.x * 256 + threadIdx.x;
    if (idx >= ETOT * NHEAD) return;
    int e = idx >> 2, h = idx & 3;
    int d = dstl[e];
    float ev = __expf(eval[idx] - unfkey(mxkey[d * NHEAD + h]));
    eval[idx] = ev;
    atomAddF(&den[d * NHEAD + h], ev);
}

// ---------------------------------------------------------------------------
// 4c) weighted scatter:  acc[dst] += xl[src] * alpha   (native f32 atomics)
// ---------------------------------------------------------------------------
__global__ void k_edge_accum(const float* __restrict__ xl, const float* __restrict__ eval,
                             const float* __restrict__ den,
                             const int* __restrict__ srcl, const int* __restrict__ dstl,
                             float* __restrict__ acc) {
    int e = blockIdx.x * 8 + (threadIdx.x >> 5);
    if (e >= ETOT) return;
    int lane = threadIdx.x & 31;
    int s = srcl[e], d = dstl[e];
    int h = lane >> 3;
    float alpha = eval[e * NHEAD + h] / den[d * NHEAD + h];
    int d0 = lane * 8;
    const float* ps = xl + (size_t)s * 256 + d0;
    float*       pd = acc + (size_t)d * 256 + d0;
    #pragma unroll
    for (int j = 0; j < 8; ++j)
        atomAddF(pd + j, ps[j] * alpha);
}

// ---------------------------------------------------------------------------
// 5) x = relu(LN(acc + gat_bias + x)); also refresh bf16 copy (wave per row)
// ---------------------------------------------------------------------------
__global__ void k_ln_relu(const float* __restrict__ acc, const float* __restrict__ bias,
                          const float* __restrict__ lng, const float* __restrict__ lnb,
                          float* __restrict__ x, unsigned short* __restrict__ xb) {
    int row  = blockIdx.x * 8 + (threadIdx.x >> 5);
    int lane = threadIdx.x & 31;
    size_t base = (size_t)row * 256;
    float v[8], s1 = 0.f, s2 = 0.f;
    #pragma unroll
    for (int j = 0; j < 8; ++j) {
        int d = j * 32 + lane;
        float t = acc[base + d] + bias[d] + x[base + d];
        v[j] = t; s1 += t; s2 += t * t;
    }
    #pragma unroll
    for (int off = 16; off > 0; off >>= 1) {
        s1 += __shfl_xor(s1, off);
        s2 += __shfl_xor(s2, off);
    }
    float mu  = s1 * (1.f / 256.f);
    float var = s2 * (1.f / 256.f) - mu * mu;
    float rs  = rsqrtf(var + 1e-5f);
    #pragma unroll
    for (int j = 0; j < 8; ++j) {
        int d = j * 32 + lane;
        float t = (v[j] - mu) * rs * lng[d] + lnb[d];
        t = t > 0.f ? t : 0.f;
        x[base + d]  = t;
        xb[base + d] = f2bf(t);
    }
}

// ---------------------------------------------------------------------------
// 6) pooling: mean / max / sum over 512 rows per graph
// ---------------------------------------------------------------------------
__global__ void k_pool(const float* __restrict__ x, float* __restrict__ comb) {
    int g = blockIdx.x, d = threadIdx.x;                 // 32 blocks x 256 thr
    const float* base = x + (size_t)g * SSZ * 256 + d;
    float sum = 0.f, mx = -__builtin_inff();
    for (int i = 0; i < SSZ; ++i) {
        float v = base[(size_t)i * 256];
        sum += v; mx = fmaxf(mx, v);
    }
    comb[g * 768 + d]       = sum * (1.f / SSZ);
    comb[g * 768 + 256 + d] = mx;
    comb[g * 768 + 512 + d] = sum;
}

// ---------------------------------------------------------------------------
// 7) MLP layer 1:  relu(LN(comb @ w1 + b1))  -> h1[32,512]
// ---------------------------------------------------------------------------
__global__ void k_mlp1(const float* __restrict__ comb, const float* __restrict__ w1,
                       const float* __restrict__ b1, const float* __restrict__ g1,
                       const float* __restrict__ bb1, float* __restrict__ h1) {
    __shared__ float sred[512];
    int g = blockIdx.x, j = threadIdx.x;                 // 512 threads
    const float* cg = comb + g * 768;
    float a = b1[j];
    for (int k = 0; k < 768; ++k) a += cg[k] * w1[(size_t)k * 512 + j];
    sred[j] = a; __syncthreads();
    for (int s = 256; s > 0; s >>= 1) { if (j < s) sred[j] += sred[j + s]; __syncthreads(); }
    float mu = sred[0] * (1.f / 512.f); __syncthreads();
    float dv = a - mu;
    sred[j] = dv * dv; __syncthreads();
    for (int s = 256; s > 0; s >>= 1) { if (j < s) sred[j] += sred[j + s]; __syncthreads(); }
    float var = sred[0] * (1.f / 512.f);
    float v = dv * rsqrtf(var + 1e-5f) * g1[j] + bb1[j];
    h1[g * 512 + j] = v > 0.f ? v : 0.f;
}

// ---------------------------------------------------------------------------
// 8) MLP layer 2 -> d_out[32,256]
// ---------------------------------------------------------------------------
__global__ void k_mlp2(const float* __restrict__ h1, const float* __restrict__ w2,
                       const float* __restrict__ b2, const float* __restrict__ g2,
                       const float* __restrict__ bb2, float* __restrict__ out) {
    __shared__ float sred[256];
    int g = blockIdx.x, j = threadIdx.x;                 // 256 threads
    const float* hg = h1 + g * 512;
    float a = b2[j];
    for (int k = 0; k < 512; ++k) a += hg[k] * w2[(size_t)k * 256 + j];
    sred[j] = a; __syncthreads();
    for (int s = 128; s > 0; s >>= 1) { if (j < s) sred[j] += sred[j + s]; __syncthreads(); }
    float mu = sred[0] * (1.f / 256.f); __syncthreads();
    float dv = a - mu;
    sred[j] = dv * dv; __syncthreads();
    for (int s = 128; s > 0; s >>= 1) { if (j < s) sred[j] += sred[j + s]; __syncthreads(); }
    float var = sred[0] * (1.f / 256.f);
    float v = dv * rsqrtf(var + 1e-5f) * g2[j] + bb2[j];
    out[g * 256 + j] = v > 0.f ? v : 0.f;
}

// ---------------------------------------------------------------------------
// host-side orchestration
// ---------------------------------------------------------------------------
extern "C" void kernel_launch(void* const* d_in, const int* in_sizes, int n_in,
                              void* d_out, int out_size, void* d_ws, size_t ws_size,
                              hipStream_t stream) {
    const float* node_emb = (const float*)d_in[0];
    const int*   subset   = (const int*)d_in[1];
    const int*   eidx     = (const int*)d_in[2];
    const int*   batch    = (const int*)d_in[3];
    const float* Wl   = (const float*)d_in[5];
    const float* bl   = (const float*)d_in[6];
    const float* Wr   = (const float*)d_in[7];
    const float* br   = (const float*)d_in[8];
    const float* att  = (const float*)d_in[9];
    const float* gb   = (const float*)d_in[10];
    const float* lng  = (const float*)d_in[11];
    const float* lnb  = (const float*)d_in[12];
    const float* w1   = (const float*)d_in[13];
    const float* b1   = (const float*)d_in[14];
    const float* g1   = (const float*)d_in[15];
    const float* bb1  = (const float*)d_in[16];
    const float* w2   = (const float*)d_in[17];
    const float* b2   = (const float*)d_in[18];
    const float* g2   = (const float*)d_in[19];
    const float* bb2  = (const float*)d_in[20];
    float* out = (float*)d_out;

    char* p = (char*)d_ws;
    auto alloc = [&](size_t bytes) -> void* {
        void* r = (void*)p;
        p += (bytes + 255) & ~(size_t)255;
        return r;
    };
    float*          x     = (float*)         alloc((size_t)NSEG * DIM * 4);
    unsigned short* xb    = (unsigned short*)alloc((size_t)NSEG * DIM * 2);
    float*          xl    = (float*)         alloc((size_t)NSEG * DIM * 4);
    float*          xr    = (float*)         alloc((size_t)NSEG * DIM * 4);
    float*          acc   = (float*)         alloc((size_t)NSEG * DIM * 4);
    int*            srcl  = (int*)           alloc((size_t)ETOT * 4);
    int*            dstl  = (int*)           alloc((size_t)ETOT * 4);
    float*          eval  = (float*)         alloc((size_t)ETOT * NHEAD * 4);
    unsigned int*   mxkey = (unsigned int*)  alloc((size_t)NSEG * NHEAD * 4);
    float*          den   = (float*)         alloc((size_t)NSEG * NHEAD * 4);
    int*            inv   = (int*)           alloc((size_t)NPG * 4);
    float*          comb  = (float*)         alloc((size_t)BGR * 768 * 4);
    float*          h1    = (float*)         alloc((size_t)BGR * 512 * 4);

    // edge relabel (layer-invariant)
    hipMemsetAsync(inv, 0xFF, (size_t)NPG * 4, stream);
    k_build_inv<<<1, SSZ, 0, stream>>>(subset, inv);
    k_relabel<<<(ETOT + 255) / 256, 256, 0, stream>>>(eidx, batch, inv, srcl, dstl);
    k_gather<<<(NSEG * DIM) / 256, 256, 0, stream>>>(node_emb, subset, x, xb);

    for (int l = 0; l < 2; ++l) {
        const size_t wo = (size_t)l * DIM * DIM;
        const size_t vo = (size_t)l * DIM;
        dim3 ggrid(NSEG / 128, DIM / 64, 2);
        k_gemm<<<ggrid, 256, 0, stream>>>(xb, Wl + wo, Wr + wo, bl + vo, br + vo, xl, xr);

        hipMemsetAsync(mxkey, 0, (size_t)NSEG * NHEAD * 4, stream);
        hipMemsetAsync(den,   0, (size_t)NSEG * NHEAD * 4, stream);
        hipMemsetAsync(acc,   0, (size_t)NSEG * DIM * 4,  stream);

        k_edge_logit<<<(ETOT + 7) / 8, 256, 0, stream>>>(
            xl, xr, att + (size_t)l * NHEAD * CPH, srcl, dstl, eval, mxkey);
        k_edge_exp<<<(ETOT * NHEAD + 255) / 256, 256, 0, stream>>>(dstl, mxkey, eval, den);
        k_edge_accum<<<(ETOT + 7) / 8, 256, 0, stream>>>(xl, eval, den, srcl, dstl, acc);

        k_ln_relu<<<NSEG / 8, 256, 0, stream>>>(acc, gb + vo, lng + vo, lnb + vo, x, xb);
    }

    k_pool<<<BGR, 256, 0, stream>>>(x, comb);
    k_mlp1<<<BGR, 512, 0, stream>>>(comb, w1, b1, g1, bb1, h1);
    k_mlp2<<<BGR, 256, 0, stream>>>(h1, w2, b2, g2, bb2, out);
}
